// ResidualBlock_80436147519717
// MI455X (gfx1250) — matrix operand order repaired and checked by reading, not compiled, not run on hardware
//
#include <hip/hip_runtime.h>
#include <hip/hip_bf16.h>

// ---------------------------------------------------------------------------
// Types for WMMA
// ---------------------------------------------------------------------------
typedef float        v8f   __attribute__((ext_vector_type(8)));
typedef unsigned int v4u   __attribute__((ext_vector_type(4)));
typedef int          v4i   __attribute__((ext_vector_type(4)));
typedef __bf16       v16bf __attribute__((ext_vector_type(16)));

struct BFrag { v4u q0, q1; };   // 32 bytes == one 16x32 bf16 fragment per lane
static_assert(sizeof(BFrag) == 32, "frag size");

__device__ __forceinline__ unsigned short f32_to_bf16(float f) {
    unsigned int u = __float_as_uint(f);
    u += 0x7FFFu + ((u >> 16) & 1u);      // round-to-nearest-even
    return (unsigned short)(u >> 16);
}

// Problem constants: N=64, C=128, H=W=64
#define INV_M         (1.0f / 262144.0f)  // 1/(N*H*W)
#define THRESH_SUM    (0.2987f * 4096.0f) // threshold * H*W (compare sums)

// CDNA5 async global->LDS path (ASYNCcnt), with safe fallback
#if defined(__HIP_DEVICE_COMPILE__) && __has_builtin(__builtin_amdgcn_global_load_async_to_lds_b128)
#define RB_ASYNC_LDS 1
#else
#define RB_ASYNC_LDS 0
#endif

__device__ __forceinline__ void rb_wait_async0() {
#if __has_builtin(__builtin_amdgcn_s_wait_asynccnt)
    __builtin_amdgcn_s_wait_asynccnt(0);
#else
    asm volatile("s_wait_asynccnt 0x0" ::: "memory");
#endif
}

__device__ __forceinline__ void rb_sched_fence() {
#if __has_builtin(__builtin_amdgcn_sched_barrier)
    __builtin_amdgcn_sched_barrier(0);    // nothing may cross this point
#endif
}

// ---------------------------------------------------------------------------
// 0) zero a float buffer (stats are atomically accumulated each call)
// ---------------------------------------------------------------------------
__global__ void rb_zero_f32(float* __restrict__ p, int n) {
    int i = blockIdx.x * 256 + threadIdx.x;
    if (i < n) p[i] = 0.0f;
}

// ---------------------------------------------------------------------------
// 1) NCHW f32 -> NHWC bf16, LDS-tiled transpose (stride 65 avoids conflicts)
//    grid = N*H blocks, 256 threads
// ---------------------------------------------------------------------------
__global__ __launch_bounds__(256) void rb_pack_x(const float* __restrict__ x,
                                                 unsigned short* __restrict__ xb) {
    __shared__ float tile[128 * 65];
    const int bx = blockIdx.x;            // n*64 + h
    const int n = bx >> 6, h = bx & 63;
    const int t = threadIdx.x;
    #pragma unroll
    for (int i = 0; i < 32; ++i) {
        int e = t + 256 * i;              // e = c*64 + w
        int c = e >> 6, w = e & 63;
        tile[c * 65 + w] = x[(((size_t)n * 128 + c) * 64 + h) * 64 + w];
    }
    __syncthreads();
    const size_t base = (size_t)bx * 8192;
    #pragma unroll
    for (int i = 0; i < 32; ++i) {
        int e = t + 256 * i;              // e = w*128 + c
        int w = e >> 7, c = e & 127;
        xb[base + e] = f32_to_bf16(tile[c * 65 + w]);
    }
}

// ---------------------------------------------------------------------------
// 2) Pack weights [O,I,3,3] f32 -> WMMA B-fragment order bf16:
//    wp[tap][t8][s][lane][e]  with  co = t8*16 + lane%16,
//    ci = s*32 + (e%8) + 8*(lane/16) + 16*(e/8)
// ---------------------------------------------------------------------------
__global__ void rb_pack_w(const float* __restrict__ w,
                          unsigned short* __restrict__ wp) {
    int g = blockIdx.x * 256 + threadIdx.x;       // < 9*8*4*32*16 = 147456
    int tap  = g >> 14;
    int t8   = (g >> 11) & 7;
    int s    = (g >> 9) & 3;
    int lane = (g >> 4) & 31;
    int e    = g & 15;
    int co = (t8 << 4) + (lane & 15);
    int ci = (s << 5) + (e & 7) + ((lane >> 4) << 3) + ((e >> 3) << 4);
    int kh = tap / 3, kw = tap % 3;
    wp[g] = f32_to_bf16(w[(((size_t)co * 128 + ci) * 3 + kh) * 3 + kw]);
}

// ---------------------------------------------------------------------------
// 3) 3x3 conv, implicit GEMM with v_wmma_f32_16x16x32_bf16.
//    Block = 256 thr (8 waves). M=128 spatial rows x N=128 output channels.
//    Wave w: M-strip [16w,16w+16) x 128 ch -> 8 v8f accumulators.
//    Weight slab (32KB/tap) streamed via GLOBAL_LOAD_ASYNC_TO_LDS_B128 into a
//    double-buffered LDS slab (overlaps WMMA), fallback: VGPR staging.
//    grid = N * H/2 = 2048 blocks.
// ---------------------------------------------------------------------------
#if RB_ASYNC_LDS
__device__ __forceinline__ void rb_async_slab(const v4u* wq, int tap, v4u* dst, int tid) {
    #pragma unroll
    for (int k = 0; k < 8; ++k) {
        const v4u* gp = wq + (size_t)tap * 2048 + tid + 256 * k;
        v4u* lp = dst + tid + 256 * k;
        __builtin_amdgcn_global_load_async_to_lds_b128(
            (__attribute__((address_space(1))) v4i*)gp,
            (__attribute__((address_space(3))) v4i*)lp,
            0, 0);
    }
}
#endif

__global__ __launch_bounds__(256, 1) void rb_conv3x3_wmma(
    const unsigned short* __restrict__ in,     // NHWC bf16 [64,64,64,128]
    const unsigned short* __restrict__ wpack,  // packed fragments
    float* __restrict__ outp,                  // NHWC f32
    float* __restrict__ gs1, float* __restrict__ gs2) {
#if RB_ASYNC_LDS
    __shared__ v4u wsh[2][2048];               // 2 x 32KB double buffer
#else
    __shared__ v4u wsh[1][2048];               // 32KB single buffer
#endif
    __shared__ float chs1[128], chs2[128];

    const int tid  = threadIdx.x;
    const int wave = tid >> 5, lane = tid & 31;
    const int lm = lane & 15, hl = lane >> 4;
    const int bx = blockIdx.x;
    const int n = bx >> 5;
    const int h_base = (bx & 31) << 1;
    const int m_base = wave << 4;
    const int mA = m_base + lm;                // A-matrix row for this lane
    const int w_pos = mA & 63, h_ofs = mA >> 6;

    if (tid < 128) { chs1[tid] = 0.0f; chs2[tid] = 0.0f; }

    const v4u* wq = (const v4u*)wpack;

#if !RB_ASYNC_LDS
    v4u wreg[8];
    #pragma unroll
    for (int k = 0; k < 8; ++k) wreg[k] = wq[tid + 256 * k];   // tap 0 slab
#else
    rb_async_slab(wq, 0, &wsh[0][0], tid);
#endif

    v8f acc[8] = {};

    #pragma unroll 1
    for (int tap = 0; tap < 9; ++tap) {
#if RB_ASYNC_LDS
        rb_wait_async0();                      // my slab[tap] loads done
        __syncthreads();                       // everyone's done; prev readers drained
        if (tap + 1 < 9)                       // stream next slab during WMMAs
            rb_async_slab(wq, tap + 1, &wsh[(tap + 1) & 1][0], tid);
        const v4u* buf = &wsh[tap & 1][0];
#else
        #pragma unroll
        for (int k = 0; k < 8; ++k) wsh[0][tid + 256 * k] = wreg[k];
        __syncthreads();
        if (tap + 1 < 9) {
            #pragma unroll
            for (int k = 0; k < 8; ++k)
                wreg[k] = wq[(size_t)(tap + 1) * 2048 + tid + 256 * k];
        }
        const v4u* buf = &wsh[0][0];
#endif
        const int dh = tap / 3 - 1, dw = tap % 3 - 1;
        const int hp = h_base + h_ofs + dh;
        const int wp = w_pos + dw;
        const bool valid = ((unsigned)hp < 64u) && ((unsigned)wp < 64u);
        const int hps = valid ? hp : 0, wps = valid ? wp : 0;
        const unsigned short* aptr =
            in + ((((size_t)n * 64 + hps) * 64 + wps) * 128 + 8 * hl);

        // prefetch next tap's A row (global_prefetch_b8)
        if (tap + 1 < 9) {
            const int dh2 = (tap + 1) / 3 - 1, dw2 = (tap + 1) % 3 - 1;
            const int hp2 = h_base + h_ofs + dh2;
            const int wp2 = w_pos + dw2;
            if (((unsigned)hp2 < 64u) && ((unsigned)wp2 < 64u))
                __builtin_prefetch(
                    in + ((((size_t)n * 64 + hp2) * 64 + wp2) * 128 + 8 * hl), 0, 3);
        }

        #pragma unroll
        for (int s = 0; s < 4; ++s) {
            BFrag ar = {};
            if (valid) {
                // A 16x32 bf16 layout: lane half hl picks K-subblock
                ar.q0 = *(const v4u*)(aptr + s * 32);
                ar.q1 = *(const v4u*)(aptr + s * 32 + 16);
            }
            const v16bf av = __builtin_bit_cast(v16bf, ar);

            // Preload all 8 B fragments into distinct regs (one ds_load clause).
            BFrag br[8];
            #pragma unroll
            for (int t8 = 0; t8 < 8; ++t8) {
                const v4u* bp = &buf[(((t8 << 2) + s) * 32 + lane) * 2];
                br[t8].q0 = bp[0];
                br[t8].q1 = bp[1];
            }
            // Fence: loads stay above, WMMAs below -> granular dscnt waits and
            // back-to-back WMMA issue instead of load->wait->wmma serialization.
            rb_sched_fence();
            #pragma unroll
            for (int t8 = 0; t8 < 8; ++t8) {
                acc[t8] = __builtin_amdgcn_wmma_f32_16x16x32_bf16(
                    false, av, false, __builtin_bit_cast(v16bf, br[t8]),
                    (short)0, acc[t8], false, false);
            }
        }
#if !RB_ASYNC_LDS
        __syncthreads();                       // protect wsh before next fill
#endif
    }

    // Epilogue: store NHWC f32; C/D layout: reg r -> M = r + 8*hl, N = lm
    #pragma unroll
    for (int t8 = 0; t8 < 8; ++t8) {
        const int co = (t8 << 4) + lm;
        float s1l = 0.0f, s2l = 0.0f;
        #pragma unroll
        for (int r = 0; r < 8; ++r) {
            const int m  = m_base + (hl << 3) + r;
            const int gh = h_base + (m >> 6);
            const int gw = m & 63;
            const size_t row = ((size_t)n * 64 + gh) * 64 + gw;
            const float v = acc[t8][r];
            outp[row * 128 + co] = v;
            s1l += v; s2l += v * v;
        }
        atomicAdd(&chs1[co], s1l);             // ds_add_f32
        atomicAdd(&chs2[co], s2l);
    }
    __syncthreads();
    if (tid < 128) {
        atomicAdd(&gs1[tid], chs1[tid]);       // global_atomic_add_f32
        atomicAdd(&gs2[tid], chs2[tid]);
    }
}

// ---------------------------------------------------------------------------
// 4) BN finalize: per-channel scale/bias from accumulated sums
// ---------------------------------------------------------------------------
__global__ void rb_bn_finalize(const float* __restrict__ s1, const float* __restrict__ s2,
                               const float* __restrict__ gamma, const float* __restrict__ beta,
                               float* __restrict__ sc, float* __restrict__ bi) {
    int c = threadIdx.x;                       // 128 threads
    float mean = s1[c] * INV_M;
    float var  = s2[c] * INV_M - mean * mean;
    float scale = gamma[c] * rsqrtf(var + 1e-5f);
    sc[c] = scale;
    bi[c] = beta[c] - mean * scale;
}

// ---------------------------------------------------------------------------
// 5) BN1 apply + ReLU -> act (bf16 NHWC), accumulate per-(n,c) sums for mask.
// ---------------------------------------------------------------------------
__global__ __launch_bounds__(256) void rb_act_stats(
    const float* __restrict__ pre, const float* __restrict__ sc,
    const float* __restrict__ bi, unsigned short* __restrict__ act,
    float* __restrict__ chanSum) {
    __shared__ float red[256];
    const int bx = blockIdx.x;                 // n*64 + h
    const int t = threadIdx.x;
    const int c = t & 127;
    const float scale = sc[c], bias = bi[c];
    const size_t base = (size_t)bx * 8192;
    float a = 0.0f;
    #pragma unroll
    for (int i = 0; i < 32; ++i) {
        int e = t + 256 * i;
        float y = fmaxf(pre[base + e] * scale + bias, 0.0f);
        act[base + e] = f32_to_bf16(y);
        a += y;
    }
    red[t] = a;
    __syncthreads();
    if (t < 128) atomicAdd(&chanSum[(bx >> 6) * 128 + c], red[t] + red[t + 128]);
}

// ---------------------------------------------------------------------------
// 6) Channel gating: zero channels whose (n,c) spatial mean < threshold
// ---------------------------------------------------------------------------
__global__ void rb_mask_apply(unsigned short* __restrict__ act,
                              const float* __restrict__ chanSum) {
    size_t i = (size_t)blockIdx.x * 256 + threadIdx.x;  // < 33554432
    int c = (int)(i & 127);
    int n = (int)(i >> 19);                    // 64*64*128 = 2^19 elems/image
    if (chanSum[n * 128 + c] < THRESH_SUM) act[i] = 0;
}

// ---------------------------------------------------------------------------
// 7) Final fuse: BN2 + shortcut + ReLU, NHWC f32 -> NCHW f32 via LDS tile
// ---------------------------------------------------------------------------
__global__ __launch_bounds__(256) void rb_final(
    const float* __restrict__ pre, const float* __restrict__ sc,
    const float* __restrict__ bi, const float* __restrict__ x,
    float* __restrict__ out) {
    __shared__ float tile[128 * 65];
    const int bx = blockIdx.x;
    const int n = bx >> 6, h = bx & 63;
    const int t = threadIdx.x;
    const size_t base = (size_t)bx * 8192;
    #pragma unroll
    for (int i = 0; i < 32; ++i) {
        int e = t + 256 * i;                   // e = w*128 + c (coalesced read)
        int w = e >> 7, c = e & 127;
        tile[c * 65 + w] = pre[base + e] * sc[c] + bi[c];
    }
    __syncthreads();
    #pragma unroll
    for (int i = 0; i < 32; ++i) {
        int e = t + 256 * i;                   // e = c*64 + w (coalesced write)
        int c = e >> 6, w = e & 63;
        size_t idx = (((size_t)n * 128 + c) * 64 + h) * 64 + w;
        out[idx] = fmaxf(tile[c * 65 + w] + x[idx], 0.0f);
    }
}

// ---------------------------------------------------------------------------
// Workspace layout (bytes)
// ---------------------------------------------------------------------------
static constexpr size_t OFF_XBF   = 0;                       // 67108864 B
static constexpr size_t OFF_ACT1  = 67108864;                // 67108864 B
static constexpr size_t OFF_OUT1  = 134217728;               // 134217728 B
static constexpr size_t OFF_OUT2  = 268435456;               // 134217728 B
static constexpr size_t OFF_WP1   = 402653184;               // 294912 B
static constexpr size_t OFF_WP2   = OFF_WP1 + 294912;        // 294912 B
static constexpr size_t OFF_STATS = OFF_WP2 + 294912;        // 9216 floats

extern "C" void kernel_launch(void* const* d_in, const int* in_sizes, int n_in,
                              void* d_out, int out_size, void* d_ws, size_t ws_size,
                              hipStream_t stream) {
    const float* x      = (const float*)d_in[0];
    const float* w1     = (const float*)d_in[1];
    const float* gamma1 = (const float*)d_in[2];
    const float* beta1  = (const float*)d_in[3];
    const float* w2     = (const float*)d_in[4];
    const float* gamma2 = (const float*)d_in[5];
    const float* beta2  = (const float*)d_in[6];
    float* out = (float*)d_out;

    char* ws = (char*)d_ws;
    unsigned short* xbf  = (unsigned short*)(ws + OFF_XBF);
    unsigned short* act1 = (unsigned short*)(ws + OFF_ACT1);
    float* out1 = (float*)(ws + OFF_OUT1);
    float* out2 = (float*)(ws + OFF_OUT2);
    unsigned short* wp1 = (unsigned short*)(ws + OFF_WP1);
    unsigned short* wp2 = (unsigned short*)(ws + OFF_WP2);
    float* stats   = (float*)(ws + OFF_STATS);
    float* chanSum = stats;            // 8192
    float* S1a = stats + 8192;
    float* S2a = stats + 8320;
    float* S1b = stats + 8448;
    float* S2b = stats + 8576;
    float* sc1 = stats + 8704;
    float* bi1 = stats + 8832;
    float* sc2 = stats + 8960;
    float* bi2 = stats + 9088;         // total 9216 floats

    rb_zero_f32<<<36, 256, 0, stream>>>(stats, 9216);
    rb_pack_x<<<4096, 256, 0, stream>>>(x, xbf);
    rb_pack_w<<<576, 256, 0, stream>>>(w1, wp1);
    rb_pack_w<<<576, 256, 0, stream>>>(w2, wp2);

    rb_conv3x3_wmma<<<2048, 256, 0, stream>>>(xbf, wp1, out1, S1a, S2a);
    rb_bn_finalize<<<1, 128, 0, stream>>>(S1a, S2a, gamma1, beta1, sc1, bi1);
    rb_act_stats<<<4096, 256, 0, stream>>>(out1, sc1, bi1, act1, chanSum);
    rb_mask_apply<<<131072, 256, 0, stream>>>(act1, chanSum);

    rb_conv3x3_wmma<<<2048, 256, 0, stream>>>(act1, wp2, out2, S1b, S2b);
    rb_bn_finalize<<<1, 128, 0, stream>>>(S1b, S2b, gamma2, beta2, sc2, bi2);
    rb_final<<<4096, 256, 0, stream>>>(out2, sc2, bi2, x, out);
}
